// LSTMnetwork_86071144612314
// MI455X (gfx1250) — compile-verified, matrix operand order checked by reading, not run on hardware
//
#include <hip/hip_runtime.h>

// ---------------- problem constants ----------------
constexpr int kIN  = 128;
constexpr int kH   = 512;
constexpr int kB   = 256;
constexpr int kT   = 512;
constexpr int kOUT = 2;
constexpr int kG4H = 4 * kH;   // 2048

#define USE_ASYNC_LDS 1        // GLOBAL_LOAD_ASYNC_TO_LDS_B128 direct-to-LDS path

typedef __bf16 bf16;
typedef __attribute__((ext_vector_type(16))) __bf16 v16bf;
typedef __attribute__((ext_vector_type(8)))  float  v8f;

union Frag16 { v16bf v; uint4 u[2]; };   // 16 x bf16 = 32 bytes

__device__ __forceinline__ float sigmf(float x) {
  return 1.0f / (1.0f + __expf(-x));
}
__device__ __forceinline__ float fast_tanh(float x) {
  return 2.0f / (1.0f + __expf(-2.0f * x)) - 1.0f;
}

#if USE_ASYNC_LDS
// Copy 16 bytes global -> LDS without VGPR staging (ASYNCcnt-tracked).
// Generic pointers to __shared__ carry the LDS byte offset in their low 32 bits.
__device__ __forceinline__ void async_cp16(const bf16* lds_dst, const bf16* gsrc) {
  uint32_t laddr = (uint32_t)(uintptr_t)lds_dst;
  uint64_t gaddr = (uint64_t)(uintptr_t)gsrc;
  asm volatile("global_load_async_to_lds_b128 %0, %1, off"
               :: "v"(laddr), "v"(gaddr) : "memory");
}
__device__ __forceinline__ void async_wait0() {
  asm volatile("s_wait_asynccnt 0x0" ::: "memory");
}
#endif

// ---------------- helper kernels ----------------

__global__ void cvt_f32_bf16(const float* __restrict__ src, bf16* __restrict__ dst, int n) {
  int i = blockIdx.x * blockDim.x + threadIdx.x;
  if (i < n) dst[i] = (bf16)src[i];
}

// x: [B,T,IN] f32 -> xb: [T,B,IN] bf16 (per-timestep A tiles contiguous)
__global__ void cvt_x_kernel(const float* __restrict__ x, bf16* __restrict__ xb) {
  int i = blockIdx.x * blockDim.x + threadIdx.x;
  if (i >= kB * kT * kIN) return;
  int ii = i % kIN;
  int t  = (i / kIN) % kT;
  int b  = i / (kIN * kT);
  xb[((size_t)t * kB + b) * kIN + ii] = (bf16)x[i];
}

__global__ void bias_combine(const float* __restrict__ a, const float* __restrict__ b,
                             float* __restrict__ o, int n) {
  int i = blockIdx.x * blockDim.x + threadIdx.x;
  if (i < n) o[i] = a[i] + b[i];
}

// ---------------- fused LSTM step ----------------
// G = hprev @ Whh^T + xt @ Wih^T + bias ; gates ; c,h update.
// Grid (H/32, B/64) x 256 threads (8 waves). Wave w: m-stripe = w>>1, n-tile = w&1.
// Each wave accumulates 4 gate tiles for the SAME 16x16 (m,n) patch -> wave-local epilogue.
// K is staged through LDS in chunks of 64, double-buffered, filled with
// GLOBAL_LOAD_ASYNC_TO_LDS_B128 overlapping the WMMA work on the other buffer.
template <int KIN>
__global__ __launch_bounds__(256)
void lstm_step(const bf16* __restrict__ hprev, const bf16* __restrict__ xt,
               const bf16* __restrict__ Whh,   const bf16* __restrict__ Wih,
               const float* __restrict__ bias, float* __restrict__ cst,
               bf16* __restrict__ hout) {
  __shared__ bf16 sA[2][64 * 64];        // 64 rows x 64 k            (16 KB)
  __shared__ bf16 sB[2][4][32 * 64];     // 4 gates x 32 n x 64 k     (32 KB)

  const int tid  = threadIdx.x;
  const int lane = tid & 31;
  const int wave = tid >> 5;
  const int wm   = wave >> 1;            // 0..3 : 16-row stripe
  const int wn   = wave & 1;             // 0..1 : 16-col stripe
  const int bm   = blockIdx.y * 64;      // block row base (B dim)
  const int bn   = blockIdx.x * 32;      // block col base (per-gate, H dim)

  v8f acc0 = {}, acc1 = {}, acc2 = {}, acc3 = {};

  // stage one 64-wide K chunk of A and all four gate B tiles into LDS buffer `buf`
  auto fill = [&](const bf16* A, int sa, const bf16* W, int sw, int k0, int buf) {
#pragma unroll
    for (int j = 0; j < 2; ++j) {        // A: 64 rows x 64 k = 512 x 16B
      int idx = tid + j * 256;
      int row = idx >> 3;
      int ko  = (idx & 7) * 8;
#if USE_ASYNC_LDS
      async_cp16(&sA[buf][row * 64 + ko], &A[(size_t)(bm + row) * sa + k0 + ko]);
#else
      *(uint4*)&sA[buf][row * 64 + ko] =
          *(const uint4*)&A[(size_t)(bm + row) * sa + k0 + ko];
#endif
    }
#pragma unroll
    for (int j = 0; j < 4; ++j) {        // B: 4 gates x 32 n x 64 k = 1024 x 16B
      int idx = tid + j * 256;
      int g   = idx >> 8;
      int r   = idx & 255;
      int col = r >> 3;
      int ko  = (r & 7) * 8;
#if USE_ASYNC_LDS
      async_cp16(&sB[buf][g][col * 64 + ko],
                 &W[(size_t)(g * kH + bn + col) * sw + k0 + ko]);
#else
      *(uint4*)&sB[buf][g][col * 64 + ko] =
          *(const uint4*)&W[(size_t)(g * kH + bn + col) * sw + k0 + ko];
#endif
    }
  };

  // 8 WMMAs per K-chunk per wave (A fragment reused across the four gate tiles)
  auto compute = [&](int buf) {
    const int kp = (lane >> 4) * 8;      // documented 16-bit A/B lane layout
#pragma unroll
    for (int ks = 0; ks < 64; ks += 32) {
      Frag16 a, b0, b1, b2, b3;
      const bf16* pa = &sA[buf][(wm * 16 + (lane & 15)) * 64 + ks + kp];
      a.u[0] = *(const uint4*)pa;
      a.u[1] = *(const uint4*)(pa + 16);
      const int nrow = (wn * 16 + (lane & 15)) * 64 + ks + kp;
      const bf16* q0 = &sB[buf][0][nrow];
      const bf16* q1 = &sB[buf][1][nrow];
      const bf16* q2 = &sB[buf][2][nrow];
      const bf16* q3 = &sB[buf][3][nrow];
      b0.u[0] = *(const uint4*)q0; b0.u[1] = *(const uint4*)(q0 + 16);
      b1.u[0] = *(const uint4*)q1; b1.u[1] = *(const uint4*)(q1 + 16);
      b2.u[0] = *(const uint4*)q2; b2.u[1] = *(const uint4*)(q2 + 16);
      b3.u[0] = *(const uint4*)q3; b3.u[1] = *(const uint4*)(q3 + 16);
      acc0 = __builtin_amdgcn_wmma_f32_16x16x32_bf16(false, a.v, false, b0.v, (short)0, acc0, false, false);
      acc1 = __builtin_amdgcn_wmma_f32_16x16x32_bf16(false, a.v, false, b1.v, (short)0, acc1, false, false);
      acc2 = __builtin_amdgcn_wmma_f32_16x16x32_bf16(false, a.v, false, b2.v, (short)0, acc2, false, false);
      acc3 = __builtin_amdgcn_wmma_f32_16x16x32_bf16(false, a.v, false, b3.v, (short)0, acc3, false, false);
    }
  };

  auto sync_fill = [&]() {
#if USE_ASYNC_LDS
    async_wait0();                       // own wave's async LDS writes landed
#endif
    __syncthreads();                     // all waves' fills visible
  };

  int buf = 0;

  // ---- phase 0: recurrent  G += hprev @ Whh^T  (K = H) ----
  fill(hprev, kH, Whh, kH, 0, buf);
  sync_fill();
  constexpr int NC0 = kH / 64;
#pragma unroll 1
  for (int ch = 0; ch < NC0; ++ch) {
    if (ch + 1 < NC0) fill(hprev, kH, Whh, kH, (ch + 1) * 64, buf ^ 1);
    compute(buf);
    sync_fill();
    buf ^= 1;
  }

  // ---- phase 1: input  G += xt @ Wih^T  (K = KIN) ----
  fill(xt, KIN, Wih, KIN, 0, buf);
  sync_fill();
  constexpr int NC1 = KIN / 64;
#pragma unroll 1
  for (int ch = 0; ch < NC1; ++ch) {
    if (ch + 1 < NC1) fill(xt, KIN, Wih, KIN, (ch + 1) * 64, buf ^ 1);
    compute(buf);
    sync_fill();
    buf ^= 1;
  }

  // ---- epilogue: gates + cell/hidden update (wave-local) ----
  const int n = bn + wn * 16 + (lane & 15);              // column in [0,H)
  const float bi  = bias[0 * kH + n];
  const float bff = bias[1 * kH + n];
  const float bg  = bias[2 * kH + n];
  const float bo  = bias[3 * kH + n];
  const int mbase = bm + wm * 16 + (lane >> 4) * 8;      // C/D layout: VGPR r -> M = r (+8 hi half)
#pragma unroll
  for (int r = 0; r < 8; ++r) {
    const int m = mbase + r;
    const float gi = sigmf(acc0[r] + bi);
    const float gf = sigmf(acc1[r] + bff);
    const float gg = fast_tanh(acc2[r] + bg);
    const float go = sigmf(acc3[r] + bo);
    const size_t idx = (size_t)m * kH + n;
    const float cv = gf * cst[idx] + gi * gg;
    cst[idx]  = cv;
    hout[idx] = (bf16)(go * fast_tanh(cv));
  }
}

// ---------------- final FC: out[b][o] = h_last[b] . Wfc[o] + bfc[o] ----------------
__global__ void fc_kernel(const bf16* __restrict__ h, const float* __restrict__ Wfc,
                          const float* __restrict__ bfc, float* __restrict__ out) {
  int id = blockIdx.x * blockDim.x + threadIdx.x;
  if (id >= kB * kOUT) return;
  int b = id / kOUT, o = id % kOUT;
  float s = bfc[o];
  const bf16* hr = h + (size_t)b * kH;
  const float* wr = Wfc + (size_t)o * kH;
  for (int k = 0; k < kH; ++k) s += (float)hr[k] * wr[k];
  out[id] = s;
}

// ---------------- host launcher ----------------
extern "C" void kernel_launch(void* const* d_in, const int* in_sizes, int n_in,
                              void* d_out, int out_size, void* d_ws, size_t ws_size,
                              hipStream_t stream) {
  const float* x    = (const float*)d_in[0];
  const float* Wih0 = (const float*)d_in[1];
  const float* Whh0 = (const float*)d_in[2];
  const float* bih0 = (const float*)d_in[3];
  const float* bhh0 = (const float*)d_in[4];
  const float* Wih1 = (const float*)d_in[5];
  const float* Whh1 = (const float*)d_in[6];
  const float* bih1 = (const float*)d_in[7];
  const float* bhh1 = (const float*)d_in[8];
  const float* Wfc  = (const float*)d_in[9];
  const float* bfc  = (const float*)d_in[10];
  float* out = (float*)d_out;

  // carve workspace (256B aligned slices)
  size_t off = 0;
  auto carve = [&](size_t bytes) -> char* {
    char* p = (char*)d_ws + off;
    off += (bytes + 255) & ~(size_t)255;
    return p;
  };
  bf16* xb    = (bf16*)carve((size_t)kT * kB * kIN * sizeof(bf16));        //  33.5 MB
  bf16* hseq0 = (bf16*)carve((size_t)(kT + 1) * kB * kH * sizeof(bf16));   // 134.5 MB (slot 0 = h_{-1}=0)
  bf16* h1b   = (bf16*)carve((size_t)2 * kB * kH * sizeof(bf16));          // ping-pong layer-1 h
  float* c0   = (float*)carve((size_t)kB * kH * sizeof(float));
  float* c1   = (float*)carve((size_t)kB * kH * sizeof(float));
  bf16* wih0b = (bf16*)carve((size_t)kG4H * kIN * sizeof(bf16));
  bf16* whh0b = (bf16*)carve((size_t)kG4H * kH  * sizeof(bf16));
  bf16* wih1b = (bf16*)carve((size_t)kG4H * kH  * sizeof(bf16));
  bf16* whh1b = (bf16*)carve((size_t)kG4H * kH  * sizeof(bf16));
  float* b0   = (float*)carve((size_t)kG4H * sizeof(float));
  float* b1   = (float*)carve((size_t)kG4H * sizeof(float));

  auto blocks = [](int n) { return (n + 255) / 256; };
  cvt_f32_bf16<<<blocks(kG4H * kIN), 256, 0, stream>>>(Wih0, wih0b, kG4H * kIN);
  cvt_f32_bf16<<<blocks(kG4H * kH ), 256, 0, stream>>>(Whh0, whh0b, kG4H * kH);
  cvt_f32_bf16<<<blocks(kG4H * kH ), 256, 0, stream>>>(Wih1, wih1b, kG4H * kH);
  cvt_f32_bf16<<<blocks(kG4H * kH ), 256, 0, stream>>>(Whh1, whh1b, kG4H * kH);
  bias_combine<<<blocks(kG4H), 256, 0, stream>>>(bih0, bhh0, b0, kG4H);
  bias_combine<<<blocks(kG4H), 256, 0, stream>>>(bih1, bhh1, b1, kG4H);
  cvt_x_kernel<<<blocks(kB * kT * kIN), 256, 0, stream>>>(x, xb);

  hipMemsetAsync(hseq0, 0, (size_t)kB * kH * sizeof(bf16), stream);   // h_{-1} layer 0
  hipMemsetAsync(h1b,   0, (size_t)kB * kH * sizeof(bf16), stream);   // h_{-1} layer 1 (slot 0)
  hipMemsetAsync(c0,    0, (size_t)kB * kH * sizeof(float), stream);
  hipMemsetAsync(c1,    0, (size_t)kB * kH * sizeof(float), stream);

  const dim3 grid(kH / 32, kB / 64);   // 16 x 4 = 64 workgroups per step

  // layer 0: reads hseq0[t], writes hseq0[t+1]
  for (int t = 0; t < kT; ++t) {
    lstm_step<kIN><<<grid, 256, 0, stream>>>(
        hseq0 + (size_t)t * kB * kH,
        xb    + (size_t)t * kB * kIN,
        whh0b, wih0b, b0, c0,
        hseq0 + (size_t)(t + 1) * kB * kH);
  }

  // layer 1: input is layer-0 output at time t = hseq0[t+1]; h ping-pongs
  for (int t = 0; t < kT; ++t) {
    lstm_step<kH><<<grid, 256, 0, stream>>>(
        h1b   + (size_t)(t & 1) * kB * kH,
        hseq0 + (size_t)(t + 1) * kB * kH,
        whh1b, wih1b, b1, c1,
        h1b   + (size_t)((t + 1) & 1) * kB * kH);
  }

  // final h of layer 1 lives in slot (kT & 1) == 0
  fc_kernel<<<blocks(kB * kOUT), 256, 0, stream>>>(h1b + (size_t)(kT & 1) * kB * kH,
                                                   Wfc, bfc, out);
}
